// CrossAttention_39668317945956
// MI455X (gfx1250) — matrix operand order repaired and checked by reading, not compiled, not run on hardware
//
#include <hip/hip_runtime.h>
#include <stdint.h>

// ---------------------------------------------------------------------------
// CDNA5 (gfx1250) cross-attention.
// All matmuls: v_wmma_f32_16x16x32_bf16.  GEMM B-tiles staged to LDS by the
// Tensor Data Mover (tensor_load_to_lds, TENSORcnt), double-buffered.
// ---------------------------------------------------------------------------

typedef __attribute__((ext_vector_type(16))) __bf16 v16bf;
typedef __attribute__((ext_vector_type(8)))  float  v8f;
typedef uint32_t u32x4 __attribute__((ext_vector_type(4)));
typedef int32_t  i32x8 __attribute__((ext_vector_type(8)));
typedef int32_t  i32x4 __attribute__((ext_vector_type(4)));

union BF16Frag { v16bf v; uint16_t h[16]; uint32_t u[8]; };

__device__ __forceinline__ uint16_t f32_to_bf16(float f) {
  uint32_t x = __float_as_uint(f);
  x += 0x7fffu + ((x >> 16) & 1u);          // round-to-nearest-even
  return (uint16_t)(x >> 16);
}

__device__ __forceinline__ v8f v8f_zero() {
  v8f z = {0.f, 0.f, 0.f, 0.f, 0.f, 0.f, 0.f, 0.f};
  return z;
}

// A-operand fragment (16x32 bf16, MxK). Memory row-major [row][k], ld elems.
// ISA: lane r<16 -> row r, k in {0..7,16..23}; lane r+16 -> k in {8..15,24..31}.
__device__ __forceinline__ BF16Frag load_a_frag(const uint16_t* p, int ld, int row0, int k0) {
  const int lane = threadIdx.x & 31;
  const int r = lane & 15, hi = lane >> 4;
  const uint16_t* rp = p + (size_t)(row0 + r) * ld + k0;
  BF16Frag f;
#pragma unroll
  for (int i = 0; i < 8; ++i) {
    const int ks = (i < 4) ? (hi * 8 + 2 * i) : (16 + hi * 8 + 2 * (i - 4));
    f.u[i] = *(const uint32_t*)(rp + ks);
  }
  return f;
}

// B-operand fragment (32x16 bf16, KxN). Memory holds B^T row-major [n][k].
// ISA: lanes 0-15 -> n=lane, k in {0..7,16..23}; lanes 16-31 -> n=lane-16, k in {8..15,24..31}.
__device__ __forceinline__ BF16Frag load_b_frag(const uint16_t* p, int ld, int n0, int k0) {
  const int lane = threadIdx.x & 31;
  const int r = lane & 15, hi = lane >> 4;
  const uint16_t* rp = p + (size_t)(n0 + r) * ld + k0;
  BF16Frag f;
#pragma unroll
  for (int i = 0; i < 8; ++i) f.u[i] = *(const uint32_t*)(rp + ((i < 4) ? (hi * 8 + 2 * i) : (16 + hi * 8 + 2 * (i - 4))));
  return f;
}

// Same B fragment, but from an LDS-resident [64][32] bf16 tile.
__device__ __forceinline__ BF16Frag load_b_frag_lds(const uint16_t* bb, int n_local) {
  const int lane = threadIdx.x & 31;
  const int r = lane & 15, hi = lane >> 4;
  const uint16_t* rp = bb + (n_local + r) * 32;
  BF16Frag f;
#pragma unroll
  for (int i = 0; i < 8; ++i) f.u[i] = *(const uint32_t*)(rp + ((i < 4) ? (hi * 8 + 2 * i) : (16 + hi * 8 + 2 * (i - 4))));
  return f;
}

__device__ __forceinline__ v8f wmma_bf16(const BF16Frag& a, const BF16Frag& b, v8f c) {
  return __builtin_amdgcn_wmma_f32_16x16x32_bf16(false, a.v, false, b.v,
                                                 (short)0, c, false, false);
}

// ---------------------------------------------------------------------------
// TDM: stage a 2D bf16 tile (tile_k x tile_n rows) into LDS.
// D# per CDNA5 ISA: group0 {count=1, lds_addr,
// global_addr, type=2}; group1 {data_size=1(2B), tensor_dim0=K, tensor_dim1=N,
// tile_dim0=tile_k, tile_dim1=tile_n, tensor_dim0_stride=K}; groups 2,3 zero
// (2D tile: tile_dim2=0 => unused).  Tracked by TENSORcnt.
// ---------------------------------------------------------------------------
__device__ __forceinline__ void tdm_load_2d_bf16(const uint16_t* gsrc, uint32_t lds_off,
                                                 uint32_t tile_k, uint32_t tile_n,
                                                 uint32_t tensor_k, uint32_t tensor_n,
                                                 uint32_t row_stride_elems) {
  const uint64_t ga = (uint64_t)(uintptr_t)gsrc;
  u32x4 g0;
  g0[0] = 1u;                                                  // count=1
  g0[1] = lds_off;                                             // lds_addr (bytes)
  g0[2] = (uint32_t)ga;                                        // global_addr[31:0]
  g0[3] = (uint32_t)((ga >> 32) & 0x01FFFFFFu) | (2u << 30);   // ga[56:32] | type=2
  i32x8 g1;
  g1[0] = (int32_t)(1u << 16);                                 // data_size=1 (2 bytes)
  g1[1] = (int32_t)((tensor_k & 0xFFFFu) << 16);               // tensor_dim0[15:0]
  g1[2] = (int32_t)((tensor_k >> 16) | ((tensor_n & 0xFFFFu) << 16));
  g1[3] = (int32_t)((tensor_n >> 16) | (tile_k << 16));        // tile_dim0
  g1[4] = (int32_t)(tile_n & 0xFFFFu);                         // tile_dim1, tile_dim2=0
  g1[5] = (int32_t)row_stride_elems;                           // tensor_dim0_stride[31:0]
  g1[6] = 0;                                                   // stride[47:32]=0, d1s lo=0
  g1[7] = 0;
  i32x4 gz = {0, 0, 0, 0};                                     // groups 2,3: zero (2D)
  asm volatile("tensor_load_to_lds %0, %1, %2, %3"
               :: "s"(g0), "s"(g1), "s"(gz), "s"(gz)
               : "memory");
}

// ---------------------------------------------------------------------------
// GEMM core: block = 8 waves; wave w computes rows [m0, m0+32), cols [n0, n0+64).
// B tile (64 x 32k, shared by all waves) double-buffered in LDS via TDM;
// A fragments double-buffered in registers.
// ---------------------------------------------------------------------------
__device__ __forceinline__ void gemm_core_32x64(const uint16_t* __restrict__ A,
                                                const uint16_t* __restrict__ Bt,
                                                int K, int N, int m0, int n0,
                                                uint16_t* bsm, v8f acc[2][4]) {
  const bool w0 = (threadIdx.x >> 5) == 0;

  if (w0) {
    const uint32_t lds0 = (uint32_t)(uintptr_t)bsm;
    tdm_load_2d_bf16(Bt + (size_t)n0 * K, lds0, 32u, 64u, (uint32_t)K, (uint32_t)N,
                     (uint32_t)K);
  }
  BF16Frag a_cur0 = load_a_frag(A, K, m0, 0);
  BF16Frag a_cur1 = load_a_frag(A, K, m0 + 16, 0);
  BF16Frag a_nxt0, a_nxt1;
  if (w0) __builtin_amdgcn_s_wait_tensorcnt(0);
  __syncthreads();

  for (int k0 = 0; k0 < K; k0 += 32) {
    const int cur = (k0 >> 5) & 1;
    const bool not_last = (k0 + 32) < K;
    if (not_last) {
      if (w0) {
        const uint32_t ldsn = (uint32_t)(uintptr_t)(bsm + (cur ^ 1) * (64 * 32));
        tdm_load_2d_bf16(Bt + (size_t)n0 * K + (k0 + 32), ldsn, 32u, 64u,
                         (uint32_t)K, (uint32_t)N, (uint32_t)K);
      }
      a_nxt0 = load_a_frag(A, K, m0, k0 + 32);
      a_nxt1 = load_a_frag(A, K, m0 + 16, k0 + 32);
    }
    const uint16_t* bb = bsm + cur * (64 * 32);
#pragma unroll
    for (int t = 0; t < 4; ++t) {
      BF16Frag b = load_b_frag_lds(bb, t * 16);
      acc[0][t] = wmma_bf16(a_cur0, b, acc[0][t]);
      acc[1][t] = wmma_bf16(a_cur1, b, acc[1][t]);
    }
    if (not_last) {
      if (w0) __builtin_amdgcn_s_wait_tensorcnt(0);
      __syncthreads();
      a_cur0 = a_nxt0;
      a_cur1 = a_nxt1;
    }
  }
}

// ---------------------------------------------------------------------------
// Conversion kernels
// ---------------------------------------------------------------------------
__global__ __launch_bounds__(256) void k_convert_bf16(const float* __restrict__ src,
                                                      uint16_t* __restrict__ dst, int n) {
  int i = blockIdx.x * 256 + threadIdx.x;
  if (i < n) dst[i] = f32_to_bf16(src[i]);
}

// src: [K x N] row-major f32  ->  dst: [N x K] row-major bf16
__global__ __launch_bounds__(256) void k_transpose_bf16(const float* __restrict__ src,
                                                        uint16_t* __restrict__ dst,
                                                        int K, int N) {
  int i = blockIdx.x * 256 + threadIdx.x;
  if (i < K * N) {
    int k = i / N, n = i - k * N;
    dst[(size_t)n * K + k] = f32_to_bf16(src[i]);
  }
}

// ---------------------------------------------------------------------------
// GEMM kernels: grid (M/256, N/64), block 256 (8 waves x 32 rows each)
// ---------------------------------------------------------------------------
__global__ __launch_bounds__(256) void k_gemm_bf16(const uint16_t* __restrict__ A,
                                                   const uint16_t* __restrict__ Bt,
                                                   uint16_t* __restrict__ C,
                                                   int K, int N, float scale) {
  __shared__ uint16_t bsm[2 * 64 * 32];
  const int wave = threadIdx.x >> 5;
  const int m0 = (blockIdx.x * 8 + wave) * 32;
  const int n0 = blockIdx.y * 64;
  v8f acc[2][4] = {{v8f_zero(), v8f_zero(), v8f_zero(), v8f_zero()},
                   {v8f_zero(), v8f_zero(), v8f_zero(), v8f_zero()}};
  gemm_core_32x64(A, Bt, K, N, m0, n0, bsm, acc);
  const int lane = threadIdx.x & 31, r = lane & 15, hi = lane >> 4;
#pragma unroll
  for (int s = 0; s < 2; ++s)
#pragma unroll
    for (int t = 0; t < 4; ++t)
#pragma unroll
      for (int v = 0; v < 8; ++v)
        C[(size_t)(m0 + s * 16 + hi * 8 + v) * N + (n0 + t * 16 + r)] =
            f32_to_bf16(acc[s][t][v] * scale);
}

// V projection: logical out [4096 x 1024], scattered to Vt[b][h][d][1024] bf16
__global__ __launch_bounds__(256) void k_gemm_vt(const uint16_t* __restrict__ A,
                                                 const uint16_t* __restrict__ Bt,
                                                 uint16_t* __restrict__ Vt, int K, int N) {
  __shared__ uint16_t bsm[2 * 64 * 32];
  const int wave = threadIdx.x >> 5;
  const int m0 = (blockIdx.x * 8 + wave) * 32;
  const int n0 = blockIdx.y * 64;
  v8f acc[2][4] = {{v8f_zero(), v8f_zero(), v8f_zero(), v8f_zero()},
                   {v8f_zero(), v8f_zero(), v8f_zero(), v8f_zero()}};
  gemm_core_32x64(A, Bt, K, N, m0, n0, bsm, acc);
  const int lane = threadIdx.x & 31, r = lane & 15, hi = lane >> 4;
#pragma unroll
  for (int s = 0; s < 2; ++s)
#pragma unroll
    for (int t = 0; t < 4; ++t)
#pragma unroll
      for (int v = 0; v < 8; ++v) {
        int row = m0 + s * 16 + hi * 8 + v;   // b*1024 + j
        int col = n0 + t * 16 + r;            // h*64 + d
        int b = row >> 10, j = row & 1023;
        int h = col >> 6,  d = col & 63;
        Vt[(((size_t)(b * 16 + h)) * 64 + d) * 1024 + j] = f32_to_bf16(acc[s][t][v]);
      }
}

// Final projection: f32 out + bias
__global__ __launch_bounds__(256) void k_gemm_f32_bias(const uint16_t* __restrict__ A,
                                                       const uint16_t* __restrict__ Bt,
                                                       const float* __restrict__ bias,
                                                       float* __restrict__ C,
                                                       int K, int N) {
  __shared__ uint16_t bsm[2 * 64 * 32];
  const int wave = threadIdx.x >> 5;
  const int m0 = (blockIdx.x * 8 + wave) * 32;
  const int n0 = blockIdx.y * 64;
  v8f acc[2][4] = {{v8f_zero(), v8f_zero(), v8f_zero(), v8f_zero()},
                   {v8f_zero(), v8f_zero(), v8f_zero(), v8f_zero()}};
  gemm_core_32x64(A, Bt, K, N, m0, n0, bsm, acc);
  const int lane = threadIdx.x & 31, r = lane & 15, hi = lane >> 4;
#pragma unroll
  for (int t = 0; t < 4; ++t) {
    const int col = n0 + t * 16 + r;
    const float bv = bias[col];
#pragma unroll
    for (int s = 0; s < 2; ++s)
#pragma unroll
      for (int v = 0; v < 8; ++v)
        C[(size_t)(m0 + s * 16 + hi * 8 + v) * N + col] = acc[s][t][v] + bv;
  }
}

// ---------------------------------------------------------------------------
// Flash attention: one wave per (b, h, 16-query tile); 32-key chunks.
// Q: [b*2048][1024] bf16 (pre-scaled), K: [b*1024][1024] bf16, Vt: [b][h][64][1024]
// ---------------------------------------------------------------------------
__global__ __launch_bounds__(256) void k_attention(const uint16_t* __restrict__ Qb,
                                                   const uint16_t* __restrict__ Kb,
                                                   const uint16_t* __restrict__ Vt,
                                                   const uint8_t* __restrict__ mask,
                                                   uint16_t* __restrict__ Ab) {
  constexpr int NQ = 2048, NKV = 1024, INNER = 1024, DH = 64;
  __shared__ uint16_t pbuf_all[8 * 16 * 32];

  const int wave = threadIdx.x >> 5;
  uint16_t* pbuf = pbuf_all + wave * (16 * 32);

  const int b = blockIdx.z, h = blockIdx.y;
  const int i0 = (blockIdx.x * 8 + wave) * 16;

  const uint16_t* Qh = Qb + (size_t)b * NQ * INNER + h * DH;       // row stride INNER
  const uint16_t* Kh = Kb + (size_t)b * NKV * INNER + h * DH;      // row stride INNER
  const uint16_t* Vh = Vt + ((size_t)(b * 16 + h)) * DH * NKV;     // row stride NKV
  const uint8_t*  mk = mask + (size_t)b * NKV;

  const int lane = threadIdx.x & 31, r = lane & 15, hi = lane >> 4;

  // Q fragments are loop-invariant (d = 64 -> two K=32 steps)
  const BF16Frag qf0 = load_a_frag(Qh, INNER, i0, 0);
  const BF16Frag qf1 = load_a_frag(Qh, INNER, i0, 32);

  v8f o[4] = {v8f_zero(), v8f_zero(), v8f_zero(), v8f_zero()};
  float mrow[8], lrow[8];
#pragma unroll
  for (int v = 0; v < 8; ++v) { mrow[v] = -INFINITY; lrow[v] = 0.f; }

#pragma unroll 1
  for (int j0 = 0; j0 < NKV; j0 += 32) {
    // sim tile 16x32 as two 16x16 f32 fragments
    v8f s0 = v8f_zero(), s1 = v8f_zero();
    s0 = wmma_bf16(qf0, load_b_frag(Kh, INNER, j0, 0), s0);
    s0 = wmma_bf16(qf1, load_b_frag(Kh, INNER, j0, 32), s0);
    s1 = wmma_bf16(qf0, load_b_frag(Kh, INNER, j0 + 16, 0), s1);
    s1 = wmma_bf16(qf1, load_b_frag(Kh, INNER, j0 + 16, 32), s1);

    // hoist V fragment loads: global latency hides under the softmax VALU work
    BF16Frag vf[4];
#pragma unroll
    for (int t = 0; t < 4; ++t) vf[t] = load_b_frag(Vh, NKV, t * 16, j0);

    // per-key mask (this lane owns key columns j0+r and j0+16+r)
    const bool ok0 = mk[j0 + r] != 0;
    const bool ok1 = mk[j0 + 16 + r] != 0;
#pragma unroll
    for (int v = 0; v < 8; ++v) {
      if (!ok0) s0[v] = -3.0e38f;     // finfo(f32).min semantics, like reference
      if (!ok1) s1[v] = -3.0e38f;
    }

    // online softmax per row (rows live across the 16 lanes of a half-wave)
#pragma unroll
    for (int v = 0; v < 8; ++v) {
      float cm = fmaxf(s0[v], s1[v]);
#pragma unroll
      for (int off = 1; off < 16; off <<= 1) cm = fmaxf(cm, __shfl_xor(cm, off, 32));
      const float mn = fmaxf(mrow[v], cm);
      const float alpha = __expf(mrow[v] - mn);
      mrow[v] = mn;
      const float p0 = __expf(s0[v] - mn);
      const float p1 = __expf(s1[v] - mn);
      s0[v] = p0; s1[v] = p1;
      float ps = p0 + p1;
#pragma unroll
      for (int off = 1; off < 16; off <<= 1) ps += __shfl_xor(ps, off, 32);
      lrow[v] = lrow[v] * alpha + ps;
#pragma unroll
      for (int t = 0; t < 4; ++t) o[t][v] *= alpha;
    }

    // reshape P from C-fragment layout to A-fragment layout through LDS
    __syncthreads();
#pragma unroll
    for (int v = 0; v < 8; ++v) {
      const int m = hi * 8 + v;
      pbuf[m * 32 + r]      = f32_to_bf16(s0[v]);
      pbuf[m * 32 + 16 + r] = f32_to_bf16(s1[v]);
    }
    __syncthreads();
    BF16Frag pf;
#pragma unroll
    for (int i = 0; i < 8; ++i) {
      const int ks = (i < 4) ? (hi * 8 + 2 * i) : (16 + hi * 8 + 2 * (i - 4));
      pf.u[i] = *(const uint32_t*)&pbuf[r * 32 + ks];
    }

    // o[16 x 64] += P[16 x 32] * V[j0:j0+32, 0:64]
#pragma unroll
    for (int t = 0; t < 4; ++t) o[t] = wmma_bf16(pf, vf[t], o[t]);
  }

  // normalize and store to attn_out [b*2048][1024] bf16
#pragma unroll
  for (int v = 0; v < 8; ++v) {
    const float inv = 1.0f / lrow[v];
    const int row = i0 + hi * 8 + v;
#pragma unroll
    for (int t = 0; t < 4; ++t) {
      const int col = h * DH + t * 16 + r;
      Ab[((size_t)b * NQ + row) * INNER + col] = f32_to_bf16(o[t][v] * inv);
    }
  }
}

// ---------------------------------------------------------------------------
// Host launcher
// ---------------------------------------------------------------------------
extern "C" void kernel_launch(void* const* d_in, const int* in_sizes, int n_in,
                              void* d_out, int out_size, void* d_ws, size_t ws_size,
                              hipStream_t stream) {
  (void)in_sizes; (void)n_in; (void)out_size; (void)ws_size;

  constexpr int B = 4, NQ = 2048, NKV = 1024, QD = 1024, CD = 768, INNER = 1024;
  constexpr int MQ = B * NQ;    // 8192
  constexpr int MC = B * NKV;   // 4096
  constexpr float SCALE = 0.125f;  // 64^-0.5

  const float*   x    = (const float*)d_in[0];
  const float*   ctx  = (const float*)d_in[1];
  const uint8_t* mask = (const uint8_t*)d_in[2];
  const float*   Wq   = (const float*)d_in[3];
  const float*   Wk   = (const float*)d_in[4];
  const float*   Wv   = (const float*)d_in[5];
  const float*   Wo   = (const float*)d_in[6];
  const float*   bo   = (const float*)d_in[7];
  float*         out  = (float*)d_out;

  uint8_t* ws = (uint8_t*)d_ws;
  size_t off = 0;
  auto alloc = [&](size_t bytes) -> uint16_t* {
    uint16_t* p = (uint16_t*)(ws + off);
    off += (bytes + 255) & ~(size_t)255;
    return p;
  };
  uint16_t* xb  = alloc((size_t)MQ * QD * 2);      // x bf16
  uint16_t* cb  = alloc((size_t)MC * CD * 2);      // context bf16
  uint16_t* wqt = alloc((size_t)INNER * QD * 2);   // Wq^T  [1024][1024]
  uint16_t* wkt = alloc((size_t)INNER * CD * 2);   // Wk^T  [1024][768]
  uint16_t* wvt = alloc((size_t)INNER * CD * 2);   // Wv^T  [1024][768]
  uint16_t* wot = alloc((size_t)QD * INNER * 2);   // Wo^T  [1024][1024]
  uint16_t* Qb  = alloc((size_t)MQ * INNER * 2);   // Q (pre-scaled) [b*i][h*d]
  uint16_t* Kb  = alloc((size_t)MC * INNER * 2);   // K [b*j][h*d]
  uint16_t* Vtb = alloc((size_t)B * 16 * 64 * NKV * 2); // V^T [b][h][d][j]
  uint16_t* Ab  = alloc((size_t)MQ * INNER * 2);   // attention output bf16

  // 1) precision conversion / weight transposition
  k_convert_bf16<<<(MQ * QD + 255) / 256, 256, 0, stream>>>(x, xb, MQ * QD);
  k_convert_bf16<<<(MC * CD + 255) / 256, 256, 0, stream>>>(ctx, cb, MC * CD);
  k_transpose_bf16<<<(QD * INNER + 255) / 256, 256, 0, stream>>>(Wq, wqt, QD, INNER);
  k_transpose_bf16<<<(CD * INNER + 255) / 256, 256, 0, stream>>>(Wk, wkt, CD, INNER);
  k_transpose_bf16<<<(CD * INNER + 255) / 256, 256, 0, stream>>>(Wv, wvt, CD, INNER);
  k_transpose_bf16<<<(INNER * QD + 255) / 256, 256, 0, stream>>>(Wo, wot, INNER, QD);

  // 2) input projections (WMMA + TDM-staged B tiles)
  k_gemm_bf16<<<dim3(MQ / 256, INNER / 64), 256, 0, stream>>>(xb, wqt, Qb, QD, INNER, SCALE);
  k_gemm_bf16<<<dim3(MC / 256, INNER / 64), 256, 0, stream>>>(cb, wkt, Kb, CD, INNER, 1.0f);
  k_gemm_vt <<<dim3(MC / 256, INNER / 64), 256, 0, stream>>>(cb, wvt, Vtb, CD, INNER);

  // 3) masked flash attention (WMMA)
  k_attention<<<dim3(NQ / 128, 16, B), 256, 0, stream>>>(Qb, Kb, Vtb, mask, Ab);

  // 4) output projection + bias (WMMA, f32 out)
  k_gemm_f32_bias<<<dim3(MQ / 256, QD / 64), 256, 0, stream>>>(Ab, wot, bo, out, INNER, QD);
}
